// NNGDLOSS_82334523064742
// MI455X (gfx1250) — matrix operand order repaired and checked by reading, not compiled, not run on hardware
//
#include <hip/hip_runtime.h>
#include <hip/hip_bf16.h>
#include <math.h>

// ---------------------------------------------------------------------------
// NNGDLOSS: radar waveform cross/auto-correlation sidelobe loss.
// Direct time-domain formulation: for every lag tau in [0,N):
//   C_tau[a,b] = sum_m conj(s_a[m]) * s_b[m+tau]   (zero padded)
// cm values == W * |C_tau[a,b]|^2 / N^2 over all ordered pairs, tau=0 diag
// dropped. Mapped onto v_wmma_f32_16x16x32_f16 (f16 A/B, f32 accumulate).
// Workspace: 8 element-shifted, zero-padded f16 replicas of cos/sin tables so
// every WMMA B-fragment load is a 16B-aligned global_load_b128 for any lag.
// ---------------------------------------------------------------------------

typedef _Float16 v16h __attribute__((ext_vector_type(16)));
typedef float    v8f  __attribute__((ext_vector_type(8)));

#define MWAV 32
#define NLEN 8192
#define PADC 64
#define RN   (NLEN + PADC)            // padded row length in f16 elems (8256)
#define REPS ((size_t)MWAV * RN)      // elems per shifted replica
#define NREP 8                        // replicas (shift = tau mod 8)
#define NWG  256                      // workgroups in corr kernel
#define WPB  8                        // waves per block (256 thr, wave32)
#define TOTW (NWG * WPB)              // 2048 waves; 4 lags per wave
#define TTEMP 1e-3f
#define WSIDE 0.1f

__device__ __forceinline__ v8f wmma16(v16h a, v16h b, v8f c) {
  // D = A(16x32 f16) x B(32x16 f16) + C(16x16 f32)
  return __builtin_amdgcn_wmma_f32_16x16x32_f16(false, a, false, b,
                                                (short)0, c, false, false);
}

__device__ __forceinline__ v16h ld2x8(const _Float16* p0, const _Float16* p1) {
  union U { uint4 u[2]; v16h h; } f;
  f.u[0] = *reinterpret_cast<const uint4*>(p0);   // 8 halves, 16B aligned
  f.u[1] = *reinterpret_cast<const uint4*>(p1);
  return f.h;
}

// online-logsumexp merge (z units); empty state = (m=-1e30, s=0)
__device__ __forceinline__ void lse_merge(float& m, float& s, float om, float os) {
  float nm = fmaxf(m, om);
  s = s * __expf(m - nm) + os * __expf(om - nm);
  m = nm;
}

// ---------------------------------------------------------------------------
// Kernel 1: build 8 shifted, zero-padded f16 replicas of cos/sin of 2*pi*x.
// Replica p, waveform b, column t holds s_b[t+p] (0 beyond N).
// ---------------------------------------------------------------------------
__global__ void prep_kernel(const float* __restrict__ y,
                            _Float16* __restrict__ Sr,
                            _Float16* __restrict__ Si) {
  size_t idx = (size_t)blockIdx.x * blockDim.x + threadIdx.x;
  if (idx >= (size_t)NREP * REPS) return;
  int    p = (int)(idx / REPS);
  size_t r = idx % REPS;
  int    b = (int)(r / RN);
  int    t = (int)(r % RN);
  int    j = t + p;
  float c = 0.f, s = 0.f;
  if (j < NLEN) {
    float ang = 6.28318530717958647692f * y[(size_t)b * NLEN + j];
    __sincosf(ang, &s, &c);
  }
  Sr[idx] = (_Float16)c;
  Si[idx] = (_Float16)s;
}

// ---------------------------------------------------------------------------
// Kernel 2: per-lag 32x32 complex correlation GEMMs via WMMA + reductions.
// Wave g handles lags {g, 4095-g, 4096+g, 8191-g} -> per-wave work is exactly
// constant (4N - 16382 inner-product elements). 16 WMMA + 16 b128 loads per
// K=32 step; 12 independent accumulator chains for ILP. tau is forced into an
// SGPR (readfirstlane) so the K loop is scalar-controlled (EXEC stays all-1s
// across the WMMAs, no per-lane trip counter). Epilogue is fully branchless.
// ---------------------------------------------------------------------------
__global__ void __launch_bounds__(256)
corr_kernel(const _Float16* __restrict__ Sr,
            const _Float16* __restrict__ Si,
            float* __restrict__ partials) {
  const int lane = threadIdx.x & 31;
  const int wave = threadIdx.x >> 5;
  const int g    = blockIdx.x * WPB + wave;      // [0, 2048)

  const int aRow = lane & 15;                    // A row / B column in tile
  const int hsA  = (lane & 16) ? 8  : 0;         // A K-chunk offset (ISA table)
  const int hsB  = (lane & 16) ? 8  : 0;         // B K-chunk offset (ISA table)
  const int eRow = (lane & 16) ? 8  : 0;         // D row offset for lanes>=16

  float maxA = 0.f, sumA = 0.f, maxC = 0.f, sumC = 0.f;
  float lm = -1e30f, ls = 0.f;                   // logsumexp state (z = ac/T)
  const float scale = WSIDE / ((float)NLEN * (float)NLEN);

  const size_t aBase = (size_t)aRow * RN + hsA;  // replica 0 (unshifted)

#pragma unroll 1
  for (int k = 0; k < 4; ++k) {
    // lag for this wave; provably wave-uniform -> SGPR loop control
    int tau0 = (k & 1) ? ((k + 1) * TOTW - 1 - g) : (g + k * TOTW);
    const int tau   = __builtin_amdgcn_readfirstlane(tau0);
    const int p     = tau & 7;
    const int q     = tau & ~7;                  // multiple of 8 -> 16B aligned
    const int steps = (NLEN - tau + 31) >> 5;

    const _Float16* pAr0 = Sr + aBase;
    const _Float16* pAi0 = Si + aBase;
    const _Float16* pAr1 = pAr0 + (size_t)16 * RN;
    const _Float16* pAi1 = pAi0 + (size_t)16 * RN;
    const size_t bBase = (size_t)p * REPS + (size_t)aRow * RN + (size_t)q + hsB;
    const _Float16* pBr0 = Sr + bBase;
    const _Float16* pBi0 = Si + bBase;
    const _Float16* pBr1 = pBr0 + (size_t)16 * RN;
    const _Float16* pBi1 = pBi0 + (size_t)16 * RN;

    v8f cR[2][2] = {};   // Re(C) = ar.br + ai.bi
    v8f cP[2][2] = {};   // ar.bi
    v8f cQ[2][2] = {};   // ai.br   ; Im(C) = P - Q  (conj on A)

    for (int st = 0; st < steps; ++st) {
      const int m = st << 5;
      v16h fAr[2], fAi[2], fBr[2], fBi[2];
      fAr[0] = ld2x8(pAr0 + m, pAr0 + m + 16);
      fAi[0] = ld2x8(pAi0 + m, pAi0 + m + 16);
      fAr[1] = ld2x8(pAr1 + m, pAr1 + m + 16);
      fAi[1] = ld2x8(pAi1 + m, pAi1 + m + 16);
      fBr[0] = ld2x8(pBr0 + m, pBr0 + m + 16);
      fBi[0] = ld2x8(pBi0 + m, pBi0 + m + 16);
      fBr[1] = ld2x8(pBr1 + m, pBr1 + m + 16);
      fBi[1] = ld2x8(pBi1 + m, pBi1 + m + 16);
#pragma unroll
      for (int at = 0; at < 2; ++at)
#pragma unroll
        for (int bt = 0; bt < 2; ++bt) {
          cR[at][bt] = wmma16(fAr[at], fBr[bt], cR[at][bt]);
          cR[at][bt] = wmma16(fAi[at], fBi[bt], cR[at][bt]);
          cP[at][bt] = wmma16(fAr[at], fBi[bt], cP[at][bt]);
          cQ[at][bt] = wmma16(fAi[at], fBr[bt], cQ[at][bt]);
        }
    }

    // Branchless epilogue: classify each C element, fold into reductions.
    const bool tau0f = (tau == 0);
#pragma unroll
    for (int at = 0; at < 2; ++at)
#pragma unroll
      for (int bt = 0; bt < 2; ++bt)
#pragma unroll
        for (int e = 0; e < 8; ++e) {
          float cr = cR[at][bt][e];
          float ci = cP[at][bt][e] - cQ[at][bt][e];
          float ac = fmaf(cr, cr, ci * ci) * scale;
          // diagonal only possible when at==bt (compile-time prune)
          const bool dg   = (at == bt) && ((eRow + e) == aRow);
          const bool skip = dg && tau0f;          // zero-lag autocorr dropped
          float vA = (dg && !skip) ? ac : 0.0f;   // v_cndmask
          float vC = (!dg)         ? ac : 0.0f;
          maxA = fmaxf(maxA, vA);  sumA += vA;
          maxC = fmaxf(maxC, vC);  sumC += vC;
          float z  = skip ? -1e30f : ac * (1.0f / TTEMP);
          float nm = fmaxf(lm, z);                // branchless online LSE
          ls = ls * __expf(lm - nm) + __expf(z - nm);
          lm = nm;
        }
  }

  // wave32 cross-lane reduction
#pragma unroll
  for (int off = 16; off > 0; off >>= 1) {
    maxA = fmaxf(maxA, __shfl_xor(maxA, off));
    maxC = fmaxf(maxC, __shfl_xor(maxC, off));
    sumA += __shfl_xor(sumA, off);
    sumC += __shfl_xor(sumC, off);
    float om = __shfl_xor(lm, off);
    float os = __shfl_xor(ls, off);
    lse_merge(lm, ls, om, os);
  }

  __shared__ float red[WPB][6];
  if (lane == 0) {
    red[wave][0] = maxA; red[wave][1] = sumA;
    red[wave][2] = maxC; red[wave][3] = sumC;
    red[wave][4] = lm;   red[wave][5] = ls;
  }
  __syncthreads();
  if (threadIdx.x == 0) {
    float MA = 0.f, SA = 0.f, MC = 0.f, SC = 0.f, LM = -1e30f, LS = 0.f;
    for (int w = 0; w < WPB; ++w) {
      MA = fmaxf(MA, red[w][0]); SA += red[w][1];
      MC = fmaxf(MC, red[w][2]); SC += red[w][3];
      lse_merge(LM, LS, red[w][4], red[w][5]);
    }
    float* o = partials + (size_t)blockIdx.x * 6;
    o[0] = MA; o[1] = SA; o[2] = MC; o[3] = SC; o[4] = LM; o[5] = LS;
  }
}

// ---------------------------------------------------------------------------
// Kernel 3: merge NWG partials -> (loss, PSL, APSL, CPSL, ISL, AISL, CISL)
// ---------------------------------------------------------------------------
__global__ void finalize_kernel(const float* __restrict__ partials,
                                float* __restrict__ out) {
  if (blockIdx.x != 0 || threadIdx.x != 0) return;
  float MA = 0.f, SA = 0.f, MC = 0.f, SC = 0.f, LM = -1e30f, LS = 0.f;
  for (int i = 0; i < NWG; ++i) {
    const float* p = partials + (size_t)i * 6;
    MA = fmaxf(MA, p[0]); SA += p[1];
    MC = fmaxf(MC, p[2]); SC += p[3];
    lse_merge(LM, LS, p[4], p[5]);
  }
  float loss = TTEMP * (LM + logf(LS));   // T * logsumexp(ac / T)
  out[0] = loss;
  out[1] = fmaxf(MA, MC);   // PSL  = max(ac)
  out[2] = MA;              // APSL = max over autocorr sidelobes
  out[3] = MC;              // CPSL = max over cross-correlations
  out[4] = SA + SC;         // ISL
  out[5] = SA;              // AISL
  out[6] = SC;              // CISL
}

extern "C" void kernel_launch(void* const* d_in, const int* in_sizes, int n_in,
                              void* d_out, int out_size, void* d_ws, size_t ws_size,
                              hipStream_t stream) {
  (void)in_sizes; (void)n_in; (void)out_size; (void)ws_size;
  const float* y_pred = (const float*)d_in[0];   // [1, M*N] f32 phases
  float* out = (float*)d_out;                    // 7 f32 outputs
  char* ws = (char*)d_ws;

  // ws layout: Sr replicas | Si replicas | per-block partials (~8.46 MB total)
  _Float16* Sr = (_Float16*)ws;
  _Float16* Si = Sr + (size_t)NREP * REPS;
  float* partials = (float*)(ws + 2 * (size_t)NREP * REPS * sizeof(_Float16));

  size_t total = (size_t)NREP * REPS;
  int prep_blocks = (int)((total + 255) / 256);
  prep_kernel<<<prep_blocks, 256, 0, stream>>>(y_pred, Sr, Si);
  corr_kernel<<<NWG, 256, 0, stream>>>(Sr, Si, partials);
  finalize_kernel<<<1, 32, 0, stream>>>(partials, out);
}
